// QuantumProjectionAttention_57690000720066
// MI455X (gfx1250) — compile-verified, matrix operand order checked by reading, not run on hardware
//
#include <hip/hip_runtime.h>
#include <hip/hip_bf16.h>

typedef __attribute__((ext_vector_type(16))) _Float16 v16h;
typedef __attribute__((ext_vector_type(8)))  _Float16 v8h;
typedef __attribute__((ext_vector_type(8)))  float    v8f;

#define Bn 8
#define Ln 1024
#define Sn 1024
#define Hn 8
#define Dn 64
static constexpr float PI_F = 3.14159265358979323846f;

// ---------------- 4-qubit state sim helpers (all-constant indexing -> registers) ----

template<int MASK>
__device__ __forceinline__ void apply1q(float* sr, float* si,
    float m00r, float m00i, float m01r, float m01i,
    float m10r, float m10i, float m11r, float m11i)
{
#pragma unroll
  for (int i = 0; i < 16; ++i) {
    if (i & MASK) continue;
    const int j = i | MASK;
    const float a0r = sr[i], a0i = si[i], a1r = sr[j], a1i = si[j];
    sr[i] = m00r*a0r - m00i*a0i + m01r*a1r - m01i*a1i;
    si[i] = m00r*a0i + m00i*a0r + m01r*a1i + m01i*a1r;
    sr[j] = m10r*a0r - m10i*a0i + m11r*a1r - m11i*a1i;
    si[j] = m10r*a0i + m10i*a0r + m11r*a1i + m11i*a1r;
  }
}

template<int MASK>
__device__ __forceinline__ void applyRX(float* sr, float* si, float theta)
{
  const float c = __cosf(0.5f * theta);
  const float s = __sinf(0.5f * theta);
  // RX = [[c, -i s], [-i s, c]]
  apply1q<MASK>(sr, si, c, 0.f, 0.f, -s, 0.f, -s, c, 0.f);
}

template<int CM, int TM>
__device__ __forceinline__ void applyCNOT(float* sr, float* si)
{
#pragma unroll
  for (int i = 0; i < 16; ++i) {
    if ((i & CM) && !(i & TM)) {
      const int j = i | TM;
      float tr = sr[i]; sr[i] = sr[j]; sr[j] = tr;
      float ti = si[i]; si[i] = si[j]; si[j] = ti;
    }
  }
}

__device__ __forceinline__ void rotMat(float phi, float th, float om, float* m)
{
  const float ct = __cosf(0.5f * th), st = __sinf(0.5f * th);
  const float ap = -0.5f * (phi + om);   // ep = exp(i*ap)
  const float am =  0.5f * (phi - om);   // em = exp(i*am)
  const float epr = __cosf(ap), epi = __sinf(ap);
  const float emr = __cosf(am), emi = __sinf(am);
  m[0] =  epr*ct; m[1] =  epi*ct;   // m00 = ep*ct
  m[2] = -emr*st; m[3] = -emi*st;   // m01 = -em*st
  m[4] =  emr*st; m[5] = -emi*st;   // m10 = conj(em)*st
  m[6] =  epr*ct; m[7] = -epi*ct;   // m11 = conj(ep)*ct
}

// --------------- Stage 1: pre-proj + quantum circuit + post-proj -> f16 (B,H,L,D) ---

__global__ void qp_feat_kernel(const float* __restrict__ X,
                               const float* __restrict__ Wpre,  // (64,4)
                               const float* __restrict__ bpre,  // (4)
                               const float* __restrict__ qw,    // (1,4,3)
                               const float* __restrict__ Wpost, // (4,64)
                               const float* __restrict__ bpost, // (64)
                               _Float16* __restrict__ out)      // (B,H,L,D) f16
{
  const int n = blockIdx.x * blockDim.x + threadIdx.x;   // 0..65535 samples
  const float* x = X + (size_t)n * Dn;

  // angles = tanh(x @ Wpre + bpre) * pi
  float a0 = bpre[0], a1 = bpre[1], a2 = bpre[2], a3 = bpre[3];
  const float4* W4 = (const float4*)Wpre;
#pragma unroll 8
  for (int d = 0; d < Dn; ++d) {
    const float xd = x[d];
    const float4 w = W4[d];
    a0 = fmaf(xd, w.x, a0); a1 = fmaf(xd, w.y, a1);
    a2 = fmaf(xd, w.z, a2); a3 = fmaf(xd, w.w, a3);
  }
  a0 = tanhf(a0) * PI_F; a1 = tanhf(a1) * PI_F;
  a2 = tanhf(a2) * PI_F; a3 = tanhf(a3) * PI_F;

  // |0000> ; wire w <-> bit (3-w)
  float sr[16], si[16];
#pragma unroll
  for (int i = 0; i < 16; ++i) { sr[i] = 0.f; si[i] = 0.f; }
  sr[0] = 1.f;

  applyRX<8>(sr, si, a0);
  applyRX<4>(sr, si, a1);
  applyRX<2>(sr, si, a2);
  applyRX<1>(sr, si, a3);

  float m[8];
  rotMat(qw[0], qw[1],  qw[2],  m); apply1q<8>(sr, si, m[0],m[1],m[2],m[3],m[4],m[5],m[6],m[7]);
  rotMat(qw[3], qw[4],  qw[5],  m); apply1q<4>(sr, si, m[0],m[1],m[2],m[3],m[4],m[5],m[6],m[7]);
  rotMat(qw[6], qw[7],  qw[8],  m); apply1q<2>(sr, si, m[0],m[1],m[2],m[3],m[4],m[5],m[6],m[7]);
  rotMat(qw[9], qw[10], qw[11], m); apply1q<1>(sr, si, m[0],m[1],m[2],m[3],m[4],m[5],m[6],m[7]);

  applyCNOT<8,4>(sr, si);   // 0->1
  applyCNOT<4,2>(sr, si);   // 1->2
  applyCNOT<2,1>(sr, si);   // 2->3
  applyCNOT<1,8>(sr, si);   // 3->0

  float z0 = 0.f, z1 = 0.f, z2 = 0.f, z3 = 0.f;
#pragma unroll
  for (int i = 0; i < 16; ++i) {
    const float p = sr[i]*sr[i] + si[i]*si[i];
    z0 += (i & 8) ? -p : p;
    z1 += (i & 4) ? -p : p;
    z2 += (i & 2) ? -p : p;
    z3 += (i & 1) ? -p : p;
  }

  // post-projection -> f16 in (B,H,L,D) layout for WMMA
  const int b = n >> 13;            // /(L*H)
  const int l = (n >> 3) & (Ln - 1);
  const int h = n & (Hn - 1);
  _Float16* o = out + ((((size_t)b * Hn + h) * Ln + l) << 6);
#pragma unroll 8
  for (int d = 0; d < Dn; ++d) {
    const float y = fmaf(z0, Wpost[d],
                    fmaf(z1, Wpost[Dn + d],
                    fmaf(z2, Wpost[2*Dn + d],
                    fmaf(z3, Wpost[3*Dn + d], bpost[d]))));
    o[d] = (_Float16)y;
  }
}

// --------------- Stage 2: V transpose/convert -> f16 (B,H,D,S) ------------------

__global__ void qp_vt_kernel(const float* __restrict__ V, _Float16* __restrict__ vt)
{
  const int i = blockIdx.x * blockDim.x + threadIdx.x;    // over (bh, d, s)
  const int s  = i & (Sn - 1);
  const int d  = (i >> 10) & (Dn - 1);
  const int bh = i >> 16;
  const int b = bh >> 3, h = bh & (Hn - 1);
  vt[i] = (_Float16)V[((((size_t)b * Sn + s) * Hn) + h) * Dn + d];
}

// --------------- Stage 3: scores = scale * Q'K'^T  (WMMA f16 -> f32) ------------
// A frag (16x32 f16): lane l holds row M=l&15; K = kb+{0..7} and kb+{16..23}, kb=(l>>4)*8
// B frag (32x16 f16): lane l holds col N=l&15; same K pattern -> contiguous loads of K rows.

__global__ void qp_scores_kernel(const _Float16* __restrict__ qf,
                                 const _Float16* __restrict__ kf,
                                 float* __restrict__ attn)
{
  const int wid  = (blockIdx.x * blockDim.x + threadIdx.x) >> 5;
  const int lane = threadIdx.x & 31;
  const int st = wid & 63;            // s tile
  const int mt = (wid >> 6) & 63;     // l tile
  const int bh = wid >> 12;

  const _Float16* qrow = qf + ((size_t)bh * Ln + mt * 16 + (lane & 15)) * Dn;
  const _Float16* krow = kf + ((size_t)bh * Sn + st * 16 + (lane & 15)) * Dn;
  const int kb = (lane >> 4) * 8;

  v8f c = {};
#pragma unroll
  for (int kc = 0; kc < 2; ++kc) {
    const int o = kc * 32 + kb;
    const v8h alo = *(const v8h*)(qrow + o);
    const v8h ahi = *(const v8h*)(qrow + o + 16);
    const v8h blo = *(const v8h*)(krow + o);
    const v8h bhi = *(const v8h*)(krow + o + 16);
    v16h a, b;
#pragma unroll
    for (int i = 0; i < 8; ++i) {
      a[i] = alo[i]; a[8 + i] = ahi[i];
      b[i] = blo[i]; b[8 + i] = bhi[i];
    }
    c = __builtin_amdgcn_wmma_f32_16x16x32_f16(false, a, false, b,
                                               (short)0, c, false, false);
  }

  // D layout: VGPR v, lane l -> row M = v + (l>>4)*8, col N = l&15
  float* ob = attn + ((size_t)bh * Ln + mt * 16 + (lane >> 4) * 8) * Sn
                   + st * 16 + (lane & 15);
#pragma unroll
  for (int v = 0; v < 8; ++v)
    ob[(size_t)v * Sn] = c[v] * 0.125f;    // 1/sqrt(64)
}

// --------------- Stage 4: row softmax (wave per row, single read + single write) ---

__global__ void qp_softmax_kernel(float* __restrict__ attn)
{
  const int row  = blockIdx.x * 8 + (threadIdx.x >> 5);   // 65536 rows
  const int lane = threadIdx.x & 31;
  float* p = attn + (size_t)row * Sn;

  float v[32];
  float mx = -3.4e38f;
#pragma unroll
  for (int j = 0; j < 32; ++j) { v[j] = p[lane + j * 32]; mx = fmaxf(mx, v[j]); }
#pragma unroll
  for (int off = 16; off; off >>= 1) mx = fmaxf(mx, __shfl_xor(mx, off, 32));
  float sum = 0.f;
#pragma unroll
  for (int j = 0; j < 32; ++j) { v[j] = __expf(v[j] - mx); sum += v[j]; }
#pragma unroll
  for (int off = 16; off; off >>= 1) sum += __shfl_xor(sum, off, 32);
  const float inv = 1.f / sum;
#pragma unroll
  for (int j = 0; j < 32; ++j) p[lane + j * 32] = v[j] * inv;
}

// --------------- Stage 5: context = attn @ V  (WMMA, A cvt f32->f16 on the fly) ---

__global__ void qp_context_kernel(const float* __restrict__ attn,
                                  const _Float16* __restrict__ vt,  // (B,H,D,S) f16
                                  float* __restrict__ ctx)          // (B,L,H,D) f32
{
  const int wid  = (blockIdx.x * blockDim.x + threadIdx.x) >> 5;
  const int lane = threadIdx.x & 31;
  const int nt = wid & 3;             // d tile (D=64 -> 4 tiles)
  const int mt = (wid >> 2) & 63;     // l tile
  const int bh = wid >> 8;

  const float*    prow = attn + ((size_t)bh * Ln + mt * 16 + (lane & 15)) * Sn;
  const _Float16* vrow = vt   + ((size_t)bh * Dn + nt * 16 + (lane & 15)) * Sn;
  const int kb = (lane >> 4) * 8;

  v8f acc = {};
#pragma unroll 4
  for (int kc = 0; kc < 32; ++kc) {       // K = S = 1024
    const int o = kc * 32 + kb;
    const v8f plo = *(const v8f*)(prow + o);
    const v8f phi = *(const v8f*)(prow + o + 16);
    const v8h blo = *(const v8h*)(vrow + o);
    const v8h bhi = *(const v8h*)(vrow + o + 16);
    v16h a, b;
#pragma unroll
    for (int i = 0; i < 8; ++i) {
      a[i]     = (_Float16)plo[i];
      a[8 + i] = (_Float16)phi[i];
      b[i]     = blo[i];
      b[8 + i] = bhi[i];
    }
    acc = __builtin_amdgcn_wmma_f32_16x16x32_f16(false, a, false, b,
                                                 (short)0, acc, false, false);
  }

  const int b_ = bh >> 3, h = bh & (Hn - 1);
  const int d  = nt * 16 + (lane & 15);
  const int l0 = mt * 16 + (lane >> 4) * 8;
#pragma unroll
  for (int v = 0; v < 8; ++v)
    ctx[(((size_t)b_ * Ln + l0 + v) * Hn + h) * Dn + d] = acc[v];
}

// ------------------------------- launcher ----------------------------------------

extern "C" void kernel_launch(void* const* d_in, const int* in_sizes, int n_in,
                              void* d_out, int out_size, void* d_ws, size_t ws_size,
                              hipStream_t stream) {
  const float* queries   = (const float*)d_in[0];
  const float* keys      = (const float*)d_in[1];
  const float* values    = (const float*)d_in[2];
  const float* Wq_pre    = (const float*)d_in[3];
  const float* bq_pre    = (const float*)d_in[4];
  const float* Wk_pre    = (const float*)d_in[5];
  const float* bk_pre    = (const float*)d_in[6];
  const float* q_weights = (const float*)d_in[7];
  const float* k_weights = (const float*)d_in[8];
  const float* Wq_post   = (const float*)d_in[9];
  const float* bq_post   = (const float*)d_in[10];
  const float* Wk_post   = (const float*)d_in[11];
  const float* bk_post   = (const float*)d_in[12];

  float* out  = (float*)d_out;
  float* ctx  = out;                                    // (B,L,H,D) = 4194304
  float* attn = out + (size_t)Bn * Ln * Hn * Dn;        // (B,H,L,S) = 67108864

  char* ws = (char*)d_ws;
  _Float16* qf = (_Float16*)ws;                          //  8 MB (B,H,L,D) f16
  _Float16* kf = (_Float16*)(ws + ((size_t)8  << 20));   //  8 MB (B,H,S,D) f16
  _Float16* vt = (_Float16*)(ws + ((size_t)16 << 20));   //  8 MB (B,H,D,S) f16

  // 1) quantum feature maps (65536 samples each)
  qp_feat_kernel<<<256, 256, 0, stream>>>(queries, Wq_pre, bq_pre, q_weights,
                                          Wq_post, bq_post, qf);
  qp_feat_kernel<<<256, 256, 0, stream>>>(keys, Wk_pre, bk_pre, k_weights,
                                          Wk_post, bk_post, kf);
  // 2) V -> f16 (B,H,D,S)
  qp_vt_kernel<<<16384, 256, 0, stream>>>(values, vt);
  // 3) scores: 64 bh * 64*64 tiles, 8 waves per block
  qp_scores_kernel<<<32768, 256, 0, stream>>>(qf, kf, attn);
  // 4) softmax over S, wave per row
  qp_softmax_kernel<<<8192, 256, 0, stream>>>(attn);
  // 5) context: 64 bh * 64*4 tiles
  qp_context_kernel<<<2048, 256, 0, stream>>>(attn, vt, ctx);
}
